// TGCSRN_11501922418722
// MI455X (gfx1250) — matrix operand order repaired and checked by reading, not compiled, not run on hardware
//
#include <hip/hip_runtime.h>
#include <hip/hip_bf16.h>

typedef __attribute__((ext_vector_type(16))) __bf16 v16bf;
typedef __attribute__((ext_vector_type(8)))  __bf16 v8bf;
typedef __attribute__((ext_vector_type(8)))  float  v8f;

#define BB   32
#define NN   1024
#define TT   12
#define HH   64
#define CC   8
#define BN   (BB*NN)      // 32768
#define H2   128
#define H3   192

// ---------------------------------------------------------------------------
// bf16 WMMA GEMM, register-resident: C[M x 64] = A[M x K] * B[K x 64]
//   A  : row-major  (lane frag = two contiguous b128 loads)
//   BT : col-major  (lane frag = two contiguous b128 loads)
// outputs: optional f32 row-major (+bias, +=), optional bf16 row-major,
//          optional bf16 col-major (one b128 store per lane).
// 128 threads = 4 waves; each wave owns a 16x16 column block.
// grid.x = M/16, grid.y = batch.
// ---------------------------------------------------------------------------
__global__ __launch_bounds__(128)
void gemm_bf16_t(const __bf16* __restrict__ A, long long sAb, int lda,
                 const __bf16* __restrict__ BT, long long sBb, int ldbt,
                 float* __restrict__ Cf, long long sCf, int ldcf, int accumulate,
                 const float* __restrict__ bias,
                 __bf16* __restrict__ Crm, long long sCrm, int ldcrm,
                 __bf16* __restrict__ CT, long long sCT, int ldct,
                 int K)
{
    const int tid  = threadIdx.x;
    const int wave = tid >> 5, lane = tid & 31;
    const int l15  = lane & 15, half = lane >> 4;
    const int mtile = blockIdx.x * 16;
    const int col   = wave * 16 + l15;

    const __bf16* arow = A  + (long long)blockIdx.y * sAb
                            + (long long)(mtile + l15) * lda + 8 * half;
    const __bf16* bcol = BT + (long long)blockIdx.y * sBb
                            + (long long)col * ldbt + 8 * half;
    v8f acc = {};
    const int nk = K >> 5;
    for (int kt = 0; kt < nk; ++kt) {
        __builtin_prefetch(arow + 32, 0, 1);     // global_prefetch_b8
        __builtin_prefetch(bcol + 32, 0, 1);
        v8bf a0 = *(const v8bf*)(arow);
        v8bf a1 = *(const v8bf*)(arow + 16);
        v8bf b0 = *(const v8bf*)(bcol);
        v8bf b1 = *(const v8bf*)(bcol + 16);
        arow += 32; bcol += 32;
        v16bf af  = __builtin_shufflevector(a0, a1, 0,1,2,3,4,5,6,7,8,9,10,11,12,13,14,15);
        v16bf bfr = __builtin_shufflevector(b0, b1, 0,1,2,3,4,5,6,7,8,9,10,11,12,13,14,15);
        acc = __builtin_amdgcn_wmma_f32_16x16x32_bf16(
                  false, af, false, bfr, (short)0, acc, false, false);
    }
    const float bv = bias ? bias[col] : 0.0f;
    if (Cf) {
#pragma unroll
        for (int r = 0; r < 8; ++r) {            // D: M = r + 8*half, N = col
            int row = mtile + r + 8*half;
            long long o = (long long)blockIdx.y * sCf + (long long)row * ldcf + col;
            float v = acc[r] + bv;
            if (accumulate) v += Cf[o];
            Cf[o] = v;
        }
    }
    if (Crm) {
#pragma unroll
        for (int r = 0; r < 8; ++r) {
            int row = mtile + r + 8*half;
            Crm[(long long)blockIdx.y * sCrm + (long long)row * ldcrm + col] =
                (__bf16)(acc[r] + bv);
        }
    }
    if (CT) {
        v8bf o;
#pragma unroll
        for (int r = 0; r < 8; ++r) o[r] = (__bf16)(acc[r] + bv);
        *(v8bf*)(CT + (long long)blockIdx.y * sCT + (long long)col * ldct
                 + mtile + 8*half) = o;
    }
}

// ---------------------------------------------------------------------------
// Fused attention: cmap[b] = softmax(Q[b] * K[b]^T / 8) row-wise, bf16 out.
// Q fragments are loop-invariant registers; K^T fragments are contiguous.
// grid.x = N/16 (query tile), grid.y = B, 128 threads.
// ---------------------------------------------------------------------------
__global__ __launch_bounds__(128)
void attn_softmax(const __bf16* __restrict__ Qb, const __bf16* __restrict__ Kb,
                  __bf16* __restrict__ cmap)
{
    __shared__ __align__(16) __bf16 Lt[1024][16];   // [key][query row], 32KB
    const int b = blockIdx.y, rt = blockIdx.x * 16;
    const int tid = threadIdx.x;
    const int wave = tid >> 5, lane = tid & 31, l15 = lane & 15, half = lane >> 4;

    const __bf16* qrow = Qb + ((long long)b * NN + rt + l15) * HH + 8*half;
    v16bf afk[2];
#pragma unroll
    for (int ks = 0; ks < 2; ++ks) {
        v8bf a0 = *(const v8bf*)(qrow + ks*32);
        v8bf a1 = *(const v8bf*)(qrow + ks*32 + 16);
        afk[ks] = __builtin_shufflevector(a0, a1, 0,1,2,3,4,5,6,7,8,9,10,11,12,13,14,15);
    }
    const __bf16* Kg = Kb + (long long)b * NN * HH;
    for (int ct = wave; ct < 64; ct += 4) {
        v8f acc = {};
        const __bf16* kp = Kg + (long long)(ct*16 + l15) * HH + 8*half;
#pragma unroll
        for (int ks = 0; ks < 2; ++ks) {
            v8bf b0 = *(const v8bf*)(kp + ks*32);
            v8bf b1 = *(const v8bf*)(kp + ks*32 + 16);
            v16bf bfr = __builtin_shufflevector(b0, b1, 0,1,2,3,4,5,6,7,8,9,10,11,12,13,14,15);
            acc = __builtin_amdgcn_wmma_f32_16x16x32_bf16(
                      false, afk[ks], false, bfr, (short)0, acc, false, false);
        }
        v8bf o;
#pragma unroll
        for (int r = 0; r < 8; ++r) o[r] = (__bf16)(acc[r] * 0.125f);
        *(v8bf*)(&Lt[ct*16 + l15][8*half]) = o;     // ds_store_b128
    }
    __syncthreads();

    const int row = tid >> 3, sub = tid & 7;        // 8 lanes per query row
    float mx = -1e30f;
    for (int c = sub; c < 1024; c += 8) mx = fmaxf(mx, (float)Lt[c][row]);
    for (int m = 1; m < 8; m <<= 1) mx = fmaxf(mx, __shfl_xor(mx, m));
    float sm = 0.f;
    for (int c = sub; c < 1024; c += 8) sm += __expf((float)Lt[c][row] - mx);
    for (int m = 1; m < 8; m <<= 1) sm += __shfl_xor(sm, m);
    const float inv = 1.f / sm;
    __bf16* orow = cmap + ((long long)b * NN + rt + row) * NN;
    for (int c = sub; c < 1024; c += 8)
        orow[c] = (__bf16)(__expf((float)Lt[c][row] - mx) * inv);
}

// --------------------------- small helper kernels ---------------------------
__global__ void cluster_kernel(const float* u, const float* sm, int* cl) {
    int n = blockIdx.x * 64 + threadIdx.x;
    if (n >= NN) return;
    int best = 0; float bv = -1e30f;
    for (int c = 0; c < CC; ++c) {
        float uu = fminf(fmaxf(u[n*CC + c], 1e-6f), 1.f - 1e-6f);
        float v  = sm[n*CC + c] - logf(-logf(uu));
        if (v > bv) { bv = v; best = c; }
    }
    cl[n] = best;
}

__global__ void temb_kernel(const int* tpos, float* temb) {
    int bt = blockIdx.x, i = threadIdx.x;           // grid B*T, 32 threads
    int p0 = tpos[bt*2 + 0], p1 = tpos[bt*2 + 1];
    float tt = (float)((p0 <= 4) ? p1 : 47 + p1);
    float freq = __expf(-logf(10000.f) * (float)i / 32.f);
    float arg = tt * freq;
    temb[bt*HH + i]      = cosf(arg);
    temb[bt*HH + 32 + i] = sinf(arg);
}

__global__ void cvt_bf16(const float* __restrict__ s, __bf16* __restrict__ d, long long n) {
    long long i = (long long)blockIdx.x * 256 + threadIdx.x;
    if (i < n) d[i] = (__bf16)s[i];
}
__global__ void zero_f32(float* p, long long n) {
    long long i = (long long)blockIdx.x * 256 + threadIdx.x;
    if (i < n) p[i] = 0.f;
}
// f32 (B,N,H) -> bf16 row-major + bf16 col-major (B,H,N)
__global__ void cvt_h_dual(const float* __restrict__ h, __bf16* __restrict__ hb,
                           __bf16* __restrict__ hbt) {
    long long i = (long long)blockIdx.x * 256 + threadIdx.x;
    if (i >= (long long)BN * HH) return;
    float v = h[i];
    long long bn = i >> 6; int f = (int)(i & 63);
    int b = (int)(bn >> 10), n = (int)(bn & 1023);
    hb[i] = (__bf16)v;
    hbt[((long long)b * HH + f) * NN + n] = (__bf16)v;
}
// W (K x 64) f32 -> WT (64 x K) bf16
__global__ void cvt_wT(const float* __restrict__ w, __bf16* __restrict__ wt, int K) {
    int i = blockIdx.x * 256 + threadIdx.x;
    if (i >= K * 64) return;
    int k = i >> 6, c = i & 63;
    wt[c * K + k] = (__bf16)w[i];
}
__global__ void cat_head(const __bf16* __restrict__ s, __bf16* __restrict__ cat) {
    long long i = (long long)blockIdx.x * 256 + threadIdx.x;   // over BN*64
    if (i >= (long long)BN * HH) return;
    long long row = i >> 6; int c = (int)(i & 63);
    cat[row * H3 + c] = s[i];
}

__global__ __launch_bounds__(256)
void prior_softmax(const float* __restrict__ src, const float* __restrict__ tgt,
                   __bf16* __restrict__ out, int t)
{
    __shared__ float lrow[NN];
    __shared__ float red[256];
    __shared__ float srow[16];
    int n = blockIdx.x, tid = threadIdx.x;
    if (tid < 16) srow[tid] = src[((long long)t * NN + n) * 16 + tid];
    __syncthreads();
    const float* tg = tgt + (long long)t * 16 * NN;
    for (int m = tid; m < NN; m += 256) {
        float s = 0.f;
        for (int e = 0; e < 16; ++e) s += srow[e] * tg[e*NN + m];
        lrow[m] = fmaxf(s, 0.f);
    }
    __syncthreads();
    float mx = -1e30f;
    for (int m = tid; m < NN; m += 256) mx = fmaxf(mx, lrow[m]);
    red[tid] = mx; __syncthreads();
    for (int s = 128; s > 0; s >>= 1) { if (tid < s) red[tid] = fmaxf(red[tid], red[tid+s]); __syncthreads(); }
    mx = red[0]; __syncthreads();
    float sm = 0.f;
    for (int m = tid; m < NN; m += 256) sm += __expf(lrow[m] - mx);
    red[tid] = sm; __syncthreads();
    for (int s = 128; s > 0; s >>= 1) { if (tid < s) red[tid] += red[tid+s]; __syncthreads(); }
    sm = red[0];
    float inv = 1.f / sm;
    for (int m = tid; m < NN; m += 256)
        out[(long long)n * NN + m] = (__bf16)(__expf(lrow[m] - mx) * inv);
}

__global__ __launch_bounds__(64)
void gru_kernel(const float* __restrict__ x, const float* __restrict__ fcW,
                const float* __restrict__ fcb, const float* __restrict__ temb,
                const float* __restrict__ hprev, const float* __restrict__ Wih,
                const float* __restrict__ Whh, const float* __restrict__ bih,
                const float* __restrict__ bhh, const int* __restrict__ cluster,
                float* __restrict__ lat_raw, int t)
{
    __shared__ __align__(16) float xv[H2];
    __shared__ __align__(16) float hv[HH];
    long long bn = blockIdx.x;
    int b = (int)(bn >> 10), n = (int)(bn & 1023);
    int tid = threadIdx.x;
    float x0 = x[(bn * TT + t) * 2 + 0];
    float x1 = x[(bn * TT + t) * 2 + 1];
    xv[tid]      = x0 * fcW[tid*2] + x1 * fcW[tid*2 + 1] + fcb[tid];
    xv[64 + tid] = temb[(b*TT + t)*HH + tid];
    hv[tid]      = hprev[bn*HH + tid];
    __syncthreads();
    int c = cluster[n];
    const float4* xv4 = (const float4*)xv;
    const float4* hv4 = (const float4*)hv;
    float gi[3], gh[3];
    for (int g = 0; g < 3; ++g) {
        int row = g*HH + tid;                    // rows (tid,+64,+128) = (r,z,n)
        const float4* wi = (const float4*)(Wih + ((long long)c*H3 + row) * H2);
        const float4* wh = (const float4*)(Whh + ((long long)c*H3 + row) * HH);
        float si = bih[c*H3 + row], sh = bhh[c*H3 + row];
        for (int k = 0; k < H2/4; ++k) {
            float4 a = xv4[k], w = wi[k];
            si += a.x*w.x + a.y*w.y + a.z*w.z + a.w*w.w;
        }
        for (int k = 0; k < HH/4; ++k) {
            float4 a = hv4[k], w = wh[k];
            sh += a.x*w.x + a.y*w.y + a.z*w.z + a.w*w.w;
        }
        gi[g] = si; gh[g] = sh;
    }
    float r  = 1.f / (1.f + __expf(-(gi[0] + gh[0])));
    float z  = 1.f / (1.f + __expf(-(gi[1] + gh[1])));
    float nv = tanhf(gi[2] + r * gh[2]);
    lat_raw[bn*HH + tid] = (1.f - z) * nv + z * hv[tid];
}

__global__ __launch_bounds__(256)
void bn_stats(const float* __restrict__ latr, float* __restrict__ mv) {
    int h = blockIdx.x;
    __shared__ float s1[256], s2[256];
    float a = 0.f, b = 0.f;
    for (int i = threadIdx.x; i < BN; i += 256) {
        float v = latr[(long long)i*HH + h];
        a += v; b += v*v;
    }
    s1[threadIdx.x] = a; s2[threadIdx.x] = b; __syncthreads();
    for (int s = 128; s > 0; s >>= 1) {
        if (threadIdx.x < s) { s1[threadIdx.x] += s1[threadIdx.x+s]; s2[threadIdx.x] += s2[threadIdx.x+s]; }
        __syncthreads();
    }
    if (threadIdx.x == 0) {
        float m = s1[0] / (float)BN;
        mv[h] = m; mv[64 + h] = s2[0] / (float)BN - m*m;
    }
}

// BN apply -> lat row-major bf16 + lat col-major bf16
__global__ void bn_apply(const float* __restrict__ latr, const float* __restrict__ mv,
                         const float* __restrict__ gamma, const float* __restrict__ beta,
                         __bf16* __restrict__ latB, __bf16* __restrict__ latBT, int t)
{
    long long i = (long long)blockIdx.x * 256 + threadIdx.x;
    if (i >= (long long)BN * HH) return;
    int h = (int)(i & 63);
    float v = gamma[t*HH + h] * (latr[i] - mv[h]) * rsqrtf(mv[64 + h] + 1e-5f) + beta[t*HH + h];
    long long bn = i >> 6;
    int b = (int)(bn >> 10), n = (int)(bn & 1023);
    latB[i] = (__bf16)v;
    latBT[((long long)b * HH + h) * NN + n] = (__bf16)v;
}

// cq/ck -> transposed bf16 (B, k, h) so they serve as BT operands
__global__ void cqck_kernel(const float* __restrict__ temb, const float* __restrict__ qW,
                            const float* __restrict__ qb, const float* __restrict__ kW,
                            const float* __restrict__ kb, __bf16* __restrict__ cqT,
                            __bf16* __restrict__ ckT, int t)
{
    int idx = blockIdx.x * 256 + threadIdx.x;        // B*4096
    if (idx >= BB * HH * HH) return;
    int b = idx >> 12, i = idx & 4095;
    const float4* te = (const float4*)(temb + (b*TT + t) * HH);
    const float4* qw = (const float4*)(qW + (long long)i*HH);
    const float4* kw = (const float4*)(kW + (long long)i*HH);
    float sq = qb[i], sk = kb[i];
    for (int h = 0; h < HH/4; ++h) {
        float4 tv = te[h], aq = qw[h], ak = kw[h];
        sq += tv.x*aq.x + tv.y*aq.y + tv.z*aq.z + tv.w*aq.w;
        sk += tv.x*ak.x + tv.y*ak.y + tv.z*ak.z + tv.w*ak.w;
    }
    int h = i >> 6, k = i & 63;                      // vec idx = h*64 + k
    cqT[(long long)b*4096 + k*HH + h] = (__bf16)sq;
    ckT[(long long)b*4096 + k*HH + h] = (__bf16)sk;
}

__global__ __launch_bounds__(128)
void head_kernel(const float* __restrict__ h, const float* __restrict__ W1,
                 const float* __restrict__ b1, const float* __restrict__ W2,
                 const float* __restrict__ b2, float* __restrict__ out)
{
    __shared__ __align__(16) float hv[HH];
    __shared__ float red[128];
    long long bn = blockIdx.x;
    int tid = threadIdx.x;
    if (tid < 64) hv[tid] = h[bn*HH + tid];
    __syncthreads();
    float p = 0.f;
    if (tid < 100) {
        float s = b1[tid];
        const float4* w = (const float4*)(W1 + tid*HH);
        const float4* hv4 = (const float4*)hv;
        for (int k = 0; k < HH/4; ++k) {
            float4 a = hv4[k], ww = w[k];
            s += a.x*ww.x + a.y*ww.y + a.z*ww.z + a.w*ww.w;
        }
        p = fmaxf(s, 0.f) * W2[tid];
    }
    red[tid] = p; __syncthreads();
    for (int s = 64; s > 0; s >>= 1) { if (tid < s) red[tid] += red[tid+s]; __syncthreads(); }
    if (tid == 0) out[bn] = red[0] + b2[0];
}

// ---------------------------------------------------------------------------
extern "C" void kernel_launch(void* const* d_in, const int* in_sizes, int n_in,
                              void* d_out, int out_size, void* d_ws, size_t ws_size,
                              hipStream_t stream)
{
    (void)in_sizes; (void)n_in; (void)out_size; (void)ws_size;
    const float* x         = (const float*)d_in[0];
    const int*   t_pos     = (const int*)  d_in[1];
    const float* gumbel    = (const float*)d_in[2];
    const float* geo_graph = (const float*)d_in[3];
    const float* soft_mat  = (const float*)d_in[4];
    const float* fc_x_W    = (const float*)d_in[5];
    const float* fc_x_b    = (const float*)d_in[6];
    const float* gru_Wih   = (const float*)d_in[7];
    const float* gru_Whh   = (const float*)d_in[8];
    const float* gru_bih   = (const float*)d_in[9];
    const float* gru_bhh   = (const float*)d_in[10];
    const float* bn_gamma  = (const float*)d_in[11];
    const float* bn_beta   = (const float*)d_in[12];
    const float* prior_src = (const float*)d_in[13];
    const float* prior_tgt = (const float*)d_in[14];
    const float* bd_W      = (const float*)d_in[15];
    const float* bd_b      = (const float*)d_in[16];
    const float* geo_W     = (const float*)d_in[17];
    const float* geo_b     = (const float*)d_in[18];
    const float* cp_W      = (const float*)d_in[19];
    const float* cp_b      = (const float*)d_in[20];
    const float* q_W       = (const float*)d_in[21];
    const float* q_b       = (const float*)d_in[22];
    const float* k_W       = (const float*)d_in[23];
    const float* k_b       = (const float*)d_in[24];
    const float* fc1_W     = (const float*)d_in[25];
    const float* fc1_b     = (const float*)d_in[26];
    const float* fc2_W     = (const float*)d_in[27];
    const float* fc2_b     = (const float*)d_in[28];

    char* ws = (char*)d_ws;
    size_t off = 0;
    auto alloc = [&](size_t bytes) -> char* {
        off = (off + 255) & ~(size_t)255;
        char* p = ws + off; off += bytes; return p;
    };
    int*     cluster = (int*)    alloc(NN * 4);
    float*   temb    = (float*)  alloc((size_t)BB*TT*HH*4);
    __bf16*  geoB    = (__bf16*) alloc((size_t)NN*NN*2);
    __bf16*  geoWT   = (__bf16*) alloc((size_t)H3*HH*2);
    __bf16*  priorB  = (__bf16*) alloc((size_t)NN*NN*2);
    float*   hA      = (float*)  alloc((size_t)BN*HH*4);
    float*   hB      = (float*)  alloc((size_t)BN*HH*4);
    __bf16*  hprevB  = (__bf16*) alloc((size_t)BN*HH*2);
    __bf16*  hprevBT = (__bf16*) alloc((size_t)BN*HH*2);
    float*   lat_raw = (float*)  alloc((size_t)BN*HH*4);
    __bf16*  latB    = (__bf16*) alloc((size_t)BN*HH*2);
    __bf16*  latBT   = (__bf16*) alloc((size_t)BN*HH*2);
    float*   meanvar = (float*)  alloc(128*4);
    __bf16*  cqT     = (__bf16*) alloc((size_t)BB*HH*HH*2);
    __bf16*  ckT     = (__bf16*) alloc((size_t)BB*HH*HH*2);
    __bf16*  Qb      = (__bf16*) alloc((size_t)BN*HH*2);
    __bf16*  Kb      = (__bf16*) alloc((size_t)BN*HH*2);
    __bf16*  cmap    = (__bf16*) alloc((size_t)BB*NN*NN*2);
    __bf16*  hopT    = (__bf16*) alloc((size_t)BN*HH*2);
    __bf16*  cat     = (__bf16*) alloc((size_t)BN*H3*2);
    __bf16*  WtT     = (__bf16*) alloc((size_t)H3*HH*2);

    const long long sQ = (long long)NN * HH;   // per-batch stride, H-wide mats
    const long long sC = (long long)NN * H3;   // per-batch stride, concat
    const long long sM = (long long)NN * NN;   // per-batch stride, cmap

    // ---- one-time prep ----
    cluster_kernel<<<16, 64, 0, stream>>>(gumbel, soft_mat, cluster);
    temb_kernel<<<BB*TT, 32, 0, stream>>>(t_pos, temb);
    cvt_bf16<<<(int)(((long long)NN*NN + 255)/256), 256, 0, stream>>>(geo_graph, geoB, (long long)NN*NN);
    cvt_wT<<<(H3*HH + 255)/256, 256, 0, stream>>>(geo_W, geoWT, H3);
    zero_f32<<<(int)(((long long)BN*HH + 255)/256), 256, 0, stream>>>(hA, (long long)BN*HH);

    float* h_prev = hA;
    float* h_cur  = hB;
    const int gBNH = (int)(((long long)BN*HH + 255)/256);

    for (int t = 0; t < TT; ++t) {
        cvt_h_dual<<<gBNH, 256, 0, stream>>>(h_prev, hprevB, hprevBT);
        prior_softmax<<<NN, 256, 0, stream>>>(prior_src, prior_tgt, priorB, t);
        gru_kernel<<<BN, 64, 0, stream>>>(x, fc_x_W, fc_x_b, temb, h_prev,
                                          gru_Wih, gru_Whh, gru_bih, gru_bhh,
                                          cluster, lat_raw, t);
        bn_stats<<<HH, 256, 0, stream>>>(lat_raw, meanvar);
        bn_apply<<<gBNH, 256, 0, stream>>>(lat_raw, meanvar, bn_gamma, bn_beta, latB, latBT, t);
        cqck_kernel<<<(BB*HH*HH + 255)/256, 256, 0, stream>>>(temb, q_W, q_b, k_W, k_b, cqT, ckT, t);

        // Q = lat @ cq, K = h_prev @ ck   (batched, K=64) -> row-major bf16
        gemm_bf16_t<<<dim3(NN/16, BB), 128, 0, stream>>>(latB,   sQ, HH, cqT, (long long)HH*HH, HH,
            nullptr, 0, 0, 0, nullptr, Qb, sQ, HH, nullptr, 0, 0, HH);
        gemm_bf16_t<<<dim3(NN/16, BB), 128, 0, stream>>>(hprevB, sQ, HH, ckT, (long long)HH*HH, HH,
            nullptr, 0, 0, 0, nullptr, Kb, sQ, HH, nullptr, 0, 0, HH);
        attn_softmax<<<dim3(NN/16, BB), 128, 0, stream>>>(Qb, Kb, cmap);

        // ---- geo GCN (adj shared, input h_prev) -> h_cur (beta=0) ----
        cat_head<<<gBNH, 256, 0, stream>>>(hprevB, cat);
        gemm_bf16_t<<<dim3(NN/16, BB), 128, 0, stream>>>(geoB, 0, NN, hprevBT, sQ, NN,
            nullptr, 0, 0, 0, nullptr, cat + HH, sC, H3, hopT, sQ, NN, NN);
        gemm_bf16_t<<<dim3(NN/16, BB), 128, 0, stream>>>(geoB, 0, NN, hopT, sQ, NN,
            nullptr, 0, 0, 0, nullptr, cat + H2, sC, H3, nullptr, 0, 0, NN);
        gemm_bf16_t<<<dim3(BN/16, 1), 128, 0, stream>>>(cat, 0, H3, geoWT, 0, H3,
            h_cur, 0, HH, 0, geo_b, nullptr, 0, 0, nullptr, 0, 0, H3);

        // ---- causal GCN (adj = cmap, input h_prev) -> h_cur (+=) ----
        cat_head<<<gBNH, 256, 0, stream>>>(hprevB, cat);
        gemm_bf16_t<<<dim3(NN/16, BB), 128, 0, stream>>>(cmap, sM, NN, hprevBT, sQ, NN,
            nullptr, 0, 0, 0, nullptr, cat + HH, sC, H3, hopT, sQ, NN, NN);
        gemm_bf16_t<<<dim3(NN/16, BB), 128, 0, stream>>>(cmap, sM, NN, hopT, sQ, NN,
            nullptr, 0, 0, 0, nullptr, cat + H2, sC, H3, nullptr, 0, 0, NN);
        cvt_wT<<<(H3*HH + 255)/256, 256, 0, stream>>>(cp_W + (long long)t*H3*HH, WtT, H3);
        gemm_bf16_t<<<dim3(BN/16, 1), 128, 0, stream>>>(cat, 0, H3, WtT, 0, H3,
            h_cur, 0, HH, 1, cp_b + t*HH, nullptr, 0, 0, nullptr, 0, 0, H3);

        // ---- unbias GCN (adj = prior_dist, input lat) -> h_cur (+=) ----
        cat_head<<<gBNH, 256, 0, stream>>>(latB, cat);
        gemm_bf16_t<<<dim3(NN/16, BB), 128, 0, stream>>>(priorB, 0, NN, latBT, sQ, NN,
            nullptr, 0, 0, 0, nullptr, cat + HH, sC, H3, hopT, sQ, NN, NN);
        gemm_bf16_t<<<dim3(NN/16, BB), 128, 0, stream>>>(priorB, 0, NN, hopT, sQ, NN,
            nullptr, 0, 0, 0, nullptr, cat + H2, sC, H3, nullptr, 0, 0, NN);
        cvt_wT<<<(H3*HH + 255)/256, 256, 0, stream>>>(bd_W + (long long)t*H3*HH, WtT, H3);
        gemm_bf16_t<<<dim3(BN/16, 1), 128, 0, stream>>>(cat, 0, H3, WtT, 0, H3,
            h_cur, 0, HH, 1, bd_b + t*HH, nullptr, 0, 0, nullptr, 0, 0, H3);

        float* tmp = h_prev; h_prev = h_cur; h_cur = tmp;
    }

    head_kernel<<<BN, 128, 0, stream>>>(h_prev, fc1_W, fc1_b, fc2_W, fc2_b, (float*)d_out);
}
